// FeatureAggregator_37263136260448
// MI455X (gfx1250) — matrix operand order repaired and checked by reading, not compiled, not run on hardware
//
#include <hip/hip_runtime.h>
#include <stdint.h>

// Problem sizes (fixed by the reference)
constexpr int DIM  = 128;    // feature dim D
constexpr int CLS  = 16;     // classes C
constexpr int NSUP = 1600;   // support rows
constexpr int NQ   = 32768;  // queries
constexpr int HIDN = 128;    // hidden width

typedef __bf16 bf16;
typedef __attribute__((ext_vector_type(16))) bf16  v16bf;
typedef __attribute__((ext_vector_type(8)))  bf16  v8bf;
typedef __attribute__((ext_vector_type(8)))  float v8f;

__device__ __forceinline__ float sigmoidf_(float x) {
    return 1.0f / (1.0f + __expf(-x));
}

// ---------------------------------------------------------------------------
// Kernel 1: per-class centers.
//   sig_mean[c][d] = sigmoid(mean of support_feat_inv rows with label c)
//   pre_s  [c][d]  = mean of support_features rows with label c
//   s_half [c][d]  = (1 + sig_mean)/2   (the pooled-feature scale)
// grid = CLS blocks, DIM threads. Label test is wave-uniform (labels indexed
// by n only), so the conditional loads do not diverge.
// ---------------------------------------------------------------------------
__global__ void centers_kernel(const float* __restrict__ inv,
                               const float* __restrict__ feat,
                               const int*   __restrict__ labels,
                               float* __restrict__ sig_mean,
                               float* __restrict__ pre_s,
                               float* __restrict__ s_half)
{
    const int c = blockIdx.x;
    const int d = threadIdx.x;
    float si = 0.f, sf = 0.f;
    int cnt = 0;
    for (int n = 0; n < NSUP; ++n) {
        if (labels[n] == c) {
            si += inv[n * DIM + d];
            sf += feat[n * DIM + d];
            ++cnt;
        }
    }
    const float invc = 1.0f / (float)(cnt < 1 ? 1 : cnt);
    const float m = si * invc;
    const float p = sf * invc;
    const float sg = sigmoidf_(m);
    sig_mean[c * DIM + d] = sg;
    pre_s   [c * DIM + d] = p;
    s_half  [c * DIM + d] = 0.5f + 0.5f * sg;
}

// ---------------------------------------------------------------------------
// Kernel 2: build per-class augmented weights, stored directly in WMMA
// B-fragment order (bf16):
//   Wc_aug[c][k][n] = W1[k][n] + sig_mean[c][k]*W1[128+k][n] + pre_s[c][k]*W1[256][n]
// B layout for v_wmma_f32_16x16x32_bf16 (ISA 7.12.2, 32x16 B):
//   lane<16 : N=lane,    K = 32*kk + 0..7 & 16..23 (2 per VGPR)
//   lane>=16: N=lane-16, K = 32*kk + 8..15 & 24..31
// Memory layout: [c][nt][kk][j(0..1)][lane(0..31)][8 bf16], so the main kernel
// loads each half-fragment with one coalesced 128-bit global load.
// 16*8*4*2*32 = 32768 threads, each writing 8 bf16.
// ---------------------------------------------------------------------------
__global__ void build_wfrag_kernel(const float* __restrict__ W1,
                                   const float* __restrict__ sig_mean,
                                   const float* __restrict__ pre_s,
                                   bf16* __restrict__ wfrag)
{
    const int tid  = blockIdx.x * blockDim.x + threadIdx.x;
    const int lane = tid & 31;
    const int j    = (tid >> 5) & 1;
    const int kk   = (tid >> 6) & 3;
    const int nt   = (tid >> 8) & 7;
    const int c    = (tid >> 11) & 15;

    const int n     = 16 * nt + (lane & 15);
    const int kbase = 32 * kk + ((lane & 16) ? 8 : 0) + j * 16;
    const float wlast = W1[256 * HIDN + n];

    v8bf outv;
#pragma unroll
    for (int e = 0; e < 8; ++e) {
        const int k = kbase + e;
        const float v = W1[k * HIDN + n]
                      + sig_mean[c * DIM + k] * W1[(DIM + k) * HIDN + n]
                      + pre_s[c * DIM + k] * wlast;
        outv[e] = (bf16)v;
    }
    const size_t off = ((((size_t)c * 8 + nt) * 4 + kk) * 2 + j) * 256 + (size_t)lane * 8;
    *reinterpret_cast<v8bf*>(wfrag + off) = outv;
}

// ---------------------------------------------------------------------------
// Kernel 3: fused main pass. One block = 128 queries, 8 waves; wave mt owns a
// 16-row A tile. Per class: 32 x v_wmma_f32_16x16x32_bf16 (16x128 @ 128x128),
// then relu(+b1)*W2, half-wave shuffle reduction, sigmoid -> gate w[q,c].
// Finally out[q][d] = q[d] * (1 + sum_c w[q,c]*s_half[c][d]).
// ---------------------------------------------------------------------------
constexpr int MBLK = 128;
constexpr int QPAD = 136;   // 272-byte row stride: 16-row b128 LDS reads hit 64 distinct banks

__global__ __launch_bounds__(256)
void fused_main_kernel(const float* __restrict__ qfeat,
                       const bf16*  __restrict__ wfrag,
                       const float* __restrict__ s_half,
                       const float* __restrict__ b1,
                       const float* __restrict__ W2,
                       const float* __restrict__ b2,
                       float* __restrict__ out)
{
    __shared__ bf16  qlds[MBLK * QPAD];   // ~34 KB bf16 queries
    __shared__ float sLDS[CLS * DIM];     // 8 KB
    __shared__ float WLDS[MBLK * CLS];    // 8 KB gates

    const int tid  = threadIdx.x;
    const int lane = tid & 31;
    const int mt   = tid >> 5;            // wave id = m-tile
    const int q0   = blockIdx.x * MBLK;

    // --- stage queries (f32 -> bf16) into padded LDS, coalesced float2 reads
    for (int it = 0; it < 32; ++it) {
        const int pairIdx = tid + it * 256;           // 8192 float2 pairs
        const int row = pairIdx >> 6;
        const int col = (pairIdx & 63) * 2;
        const float2 f = *reinterpret_cast<const float2*>(
            qfeat + (size_t)(q0 + row) * DIM + col);
        qlds[row * QPAD + col]     = (bf16)f.x;
        qlds[row * QPAD + col + 1] = (bf16)f.y;
    }
    for (int it = 0; it < 8; ++it) {
        const int idx = tid + it * 256;
        sLDS[idx] = s_half[idx];
    }
    __syncthreads();

    // --- preload class-invariant A fragments (16x32 bf16 each, ISA A layout:
    //     lane<16: K 0..7 & 16..23 ; lane>=16: K 8..15 & 24..31)
    v16bf afrag[4];
    {
        const bf16* rowb = qlds + (mt * 16 + (lane & 15)) * QPAD;
#pragma unroll
        for (int kk = 0; kk < 4; ++kk) {
            const int koff = 32 * kk + ((lane & 16) ? 8 : 0);
            const v8bf lo = *reinterpret_cast<const v8bf*>(rowb + koff);
            const v8bf hi = *reinterpret_cast<const v8bf*>(rowb + koff + 16);
            v16bf a;
#pragma unroll
            for (int e = 0; e < 8; ++e) { a[e] = lo[e]; a[8 + e] = hi[e]; }
            afrag[kk] = a;
        }
    }

    // per-lane bias / W2 for this lane's N columns
    float b1n[8], w2n[8];
#pragma unroll
    for (int nt = 0; nt < 8; ++nt) {
        b1n[nt] = b1[16 * nt + (lane & 15)];
        w2n[nt] = W2[16 * nt + (lane & 15)];
    }
    const float b2s = b2[0];

    for (int c = 0; c < CLS; ++c) {
        v8f acc[8] = {};
        const bf16* wc = wfrag + (size_t)c * (8 * 4 * 2 * 256);
#pragma unroll
        for (int kk = 0; kk < 4; ++kk) {
#pragma unroll
            for (int nt = 0; nt < 8; ++nt) {
                const bf16* fb = wc + (((size_t)nt * 4 + kk) * 2) * 256 + (size_t)lane * 8;
                const v8bf lo = *reinterpret_cast<const v8bf*>(fb);
                const v8bf hi = *reinterpret_cast<const v8bf*>(fb + 256);
                v16bf b;
#pragma unroll
                for (int e = 0; e < 8; ++e) { b[e] = lo[e]; b[8 + e] = hi[e]; }
                acc[nt] = __builtin_amdgcn_wmma_f32_16x16x32_bf16(
                    false, afrag[kk], false, b, (short)0, acc[nt], false, false);
            }
        }

        // epilogue: t[row] = sum_n relu(H + b1[n]) * W2[n]
        float partial[8];
#pragma unroll
        for (int i = 0; i < 8; ++i) partial[i] = 0.f;
#pragma unroll
        for (int nt = 0; nt < 8; ++nt) {
#pragma unroll
            for (int i = 0; i < 8; ++i) {
                float h = acc[nt][i] + b1n[nt];
                h = fmaxf(h, 0.f);
                partial[i] = fmaf(h, w2n[nt], partial[i]);
            }
        }
#pragma unroll
        for (int i = 0; i < 8; ++i) {
            float v = partial[i];
            v += __shfl_xor(v, 1);
            v += __shfl_xor(v, 2);
            v += __shfl_xor(v, 4);
            v += __shfl_xor(v, 8);   // 16-lane half reduction (C layout halves)
            const float wq = sigmoidf_(v + b2s);
            if ((lane & 15) == 0) {  // lane 0 -> row i ; lane 16 -> row 8+i
                const int row = mt * 16 + i + ((lane >> 4) << 3);
                WLDS[row * CLS + c] = wq;
            }
        }
    }
    __syncthreads();

    // --- final mix: out[q][d] = q[d] * (1 + sum_c w[q,c] * s_half[c][d])
    float sreg[4][CLS];
#pragma unroll
    for (int t = 0; t < 4; ++t)
#pragma unroll
        for (int c = 0; c < CLS; ++c)
            sreg[t][c] = sLDS[c * DIM + lane + 32 * t];

#pragma unroll 1
    for (int r = 0; r < 16; ++r) {
        const int row = mt * 16 + r;
        float wr[CLS];
#pragma unroll
        for (int c = 0; c < CLS; ++c) wr[c] = WLDS[row * CLS + c];
        const float* qrow = qfeat + (size_t)(q0 + row) * DIM;
        float*       orow = out   + (size_t)(q0 + row) * DIM;
#pragma unroll
        for (int t = 0; t < 4; ++t) {
            float g = 0.f;
#pragma unroll
            for (int c = 0; c < CLS; ++c) g = fmaf(wr[c], sreg[t][c], g);
            const int d = lane + 32 * t;
            orow[d] = qrow[d] * (1.0f + g);
        }
    }
}

// ---------------------------------------------------------------------------
extern "C" void kernel_launch(void* const* d_in, const int* in_sizes, int n_in,
                              void* d_out, int out_size, void* d_ws, size_t ws_size,
                              hipStream_t stream)
{
    const float* support_feat_inv = (const float*)d_in[0];
    const float* support_features = (const float*)d_in[1];
    const float* query_features   = (const float*)d_in[2];
    const float* W1               = (const float*)d_in[3];
    const float* b1               = (const float*)d_in[4];
    const float* W2               = (const float*)d_in[5];
    const float* b2               = (const float*)d_in[6];
    const int*   support_labels   = (const int*)d_in[7];
    // d_in[8] = query_labels (unused by the reference output)
    float* out = (float*)d_out;

    char* ws = (char*)d_ws;
    float* sig_mean = (float*)(ws);                 //  8 KB
    float* pre_s    = (float*)(ws + 8192);          //  8 KB
    float* s_half   = (float*)(ws + 16384);         //  8 KB
    bf16*  wfrag    = (bf16*)(ws + 24576);          // 512 KB bf16 fragments

    centers_kernel<<<CLS, DIM, 0, stream>>>(support_feat_inv, support_features,
                                            support_labels, sig_mean, pre_s, s_half);
    build_wfrag_kernel<<<128, 256, 0, stream>>>(W1, sig_mean, pre_s, wfrag);
    fused_main_kernel<<<NQ / MBLK, 256, 0, stream>>>(query_features, wfrag, s_half,
                                                     b1, W2, b2, out);
}
